// FNC_36653250904880
// MI455X (gfx1250) — matrix operand order repaired and checked by reading, not compiled, not run on hardware
//
#include <hip/hip_runtime.h>

#define BATCH   256
#define BIT     64
#define N_DATA  200000
#define TOPF    20
#define KNEG    2500u
#define RANK_K  20000   /* 0-based rank of kth = sorted[N-HIGH] */
#define HBINS   1024
#define SC      (8.0f / 7.2f)  /* sign scale (x8) over T=0.9*sqrt(64)=7.2 */

typedef __attribute__((ext_vector_type(16))) _Float16 v16h;
typedef __attribute__((ext_vector_type(8)))  float    v8f;

// ---------------- fusion + norms ----------------
__global__ void fuse_kernel(const float* __restrict__ iA, const float* __restrict__ iB,
                            const float* __restrict__ tA, const float* __restrict__ tB,
                            float* __restrict__ fsum, float* __restrict__ anchor,
                            float* __restrict__ newfeat) {
  int b = blockIdx.x, k = threadIdx.x;           // 64 threads
  int o = b * BIT + k;
  float ia = iA[o], ib = iB[o], ta = tA[o], tb = tB[o];
  float f[4] = { (ia + ta) * 0.5f, (ia + tb) * 0.5f, (ib + ta) * 0.5f, (ib + tb) * 0.5f };
  __shared__ float s[BIT];
  __shared__ float nv[4];
  for (int q = 0; q < 4; ++q) {
    s[k] = f[q] * f[q]; __syncthreads();
    for (int st = 32; st > 0; st >>= 1) { if (k < st) s[k] += s[k + st]; __syncthreads(); }
    if (k == 0) nv[q] = 1.0f / sqrtf(s[0]);
    __syncthreads();
  }
  fsum[o]    = 0.25f * (f[0] * nv[0] + f[1] * nv[1] + f[2] * nv[2] + f[3] * nv[3]);
  anchor[o]  = f[0];
  newfeat[o] = f[0] * nv[0];
}

// ---------------- momentum memory update (last-occurrence wins) ----------------
__global__ void update_kernel(const float* __restrict__ mem, const float* __restrict__ newfeat,
                              const int* __restrict__ bidx, float* __restrict__ outmem) {
  int b = blockIdx.x, k = threadIdx.x;           // 64 threads
  int pos = bidx[b];
  for (int j = b + 1; j < BATCH; ++j) if (bidx[j] == pos) return;  // uniform exit
  float u = mem[(size_t)pos * BIT + k] * 0.4f + newfeat[b * BIT + k] * 0.6f;
  __shared__ float s[BIT];
  __shared__ float nrm;
  s[k] = u * u; __syncthreads();
  for (int st = 32; st > 0; st >>= 1) { if (k < st) s[k] += s[k + st]; __syncthreads(); }
  if (k == 0) nrm = 1.0f / sqrtf(s[0]);
  __syncthreads();
  outmem[(size_t)pos * BIT + k] = u * nrm;
}

// ---------------- sim = fsum @ memory^T via v_wmma_f32_16x16x32_f16 ----------------
__global__ __launch_bounds__(256) void gemm_kernel(const float* __restrict__ fsum,
                                                   const float* __restrict__ mem,
                                                   float* __restrict__ sim) {
  int wave = threadIdx.x >> 5;
  int lane = threadIdx.x & 31;
  int tile = blockIdx.x * 8 + wave;              // 16 btiles x 12500 ntiles
  if (tile >= 16 * 12500) return;
  int btile = tile & 15;                          // 16 consecutive tiles share ntile -> B reuse
  int ntile = tile >> 4;
  int m = lane & 15;
  int h = lane >> 4;                              // lane half
  const float* Arow = fsum + (size_t)(btile * 16 + m) * BIT;   // A: M = lane%16
  const float* Brow = mem  + (size_t)(ntile * 16 + m) * BIT;   // B col N = lane%16
  v8f c = {};
  #pragma unroll
  for (int k0 = 0; k0 < BIT; k0 += 32) {
    v16h a, bm;
    #pragma unroll
    for (int r = 0; r < 8; ++r) {
      int ka = (r < 4) ? (8 * h + 2 * r) : (16 + 8 * h + 2 * (r - 4)); // 16-bit A 16x32 layout
      a[2 * r]     = (_Float16)Arow[k0 + ka];
      a[2 * r + 1] = (_Float16)Arow[k0 + ka + 1];
      int kb = (r < 4) ? (8 * h + 2 * r) : (16 + 8 * h + 2 * (r - 4)); // B 32x16 layout, same k order as A
      bm[2 * r]     = (_Float16)Brow[k0 + kb];
      bm[2 * r + 1] = (_Float16)Brow[k0 + kb + 1];
    }
    c = __builtin_amdgcn_wmma_f32_16x16x32_f16(false, a, false, bm, (short)0, c, false, false);
  }
  int n = ntile * 16 + (lane & 15);
  size_t base = (size_t)(btile * 16 + h * 8) * N_DATA + n;   // C: vgpr r -> M = r + 8*h
  #pragma unroll
  for (int r = 0; r < 8; ++r) sim[base + (size_t)r * N_DATA] = c[r];
}

// ---------------- per-sample top-20 + kth (20000th smallest) ----------------
__global__ __launch_bounds__(256) void topk_kth_kernel(const float* __restrict__ sim,
                                                       float* __restrict__ topv, int* __restrict__ topi,
                                                       float* __restrict__ kth) {
  int b = blockIdx.x, tid = threadIdx.x;
  const float* row = sim + (size_t)b * N_DATA;
  __shared__ unsigned hist[HBINS];
  __shared__ float stv[256 * TOPF];
  __shared__ int   sti[256 * TOPF];
  __shared__ float rv[256]; __shared__ int rp[256];
  __shared__ int sb_bin; __shared__ unsigned sb_below;
  for (int i = tid; i < HBINS; i += 256) hist[i] = 0;
  __syncthreads();

  float tvr[TOPF]; int tir[TOPF];
  #pragma unroll
  for (int j = 0; j < TOPF; ++j) { tvr[j] = -3.0e38f; tir[j] = -1; }
  float minv = -3.0e38f; int minp = 0;

  for (int n = tid; n < N_DATA; n += 256) {
    float v = row[n];
    int bin = (int)((v + 1.0f) * (HBINS * 0.5f));
    bin = bin < 0 ? 0 : (bin > HBINS - 1 ? HBINS - 1 : bin);
    atomicAdd(&hist[bin], 1u);
    if (v > minv) {
      #pragma unroll
      for (int j = 0; j < TOPF; ++j) if (j == minp) { tvr[j] = v; tir[j] = n; }
      minv = tvr[0]; minp = 0;
      #pragma unroll
      for (int j = 1; j < TOPF; ++j) if (tvr[j] < minv) { minv = tvr[j]; minp = j; }
    }
  }
  #pragma unroll
  for (int j = 0; j < TOPF; ++j) { stv[tid * TOPF + j] = tvr[j]; sti[tid * TOPF + j] = tir[j]; }
  __syncthreads();

  // coarse bin containing rank RANK_K (counting from the bottom)
  if (tid == 0) {
    unsigned c = 0; int bb = 0;
    for (; bb < HBINS - 1; ++bb) { if (c + hist[bb] > (unsigned)RANK_K) break; c += hist[bb]; }
    sb_bin = bb; sb_below = c;
  }
  __syncthreads();
  float w  = 2.0f / HBINS;
  float lo = sb_bin * w - 1.0f;
  for (int i = tid; i < HBINS; i += 256) hist[i] = 0;
  __syncthreads();
  for (int n = tid; n < N_DATA; n += 256) {
    float v = row[n];
    if (v >= lo && v < lo + w) {
      int sb = (int)((v - lo) * ((float)HBINS / w));
      sb = sb < 0 ? 0 : (sb > HBINS - 1 ? HBINS - 1 : sb);
      atomicAdd(&hist[sb], 1u);
    }
  }
  __syncthreads();
  if (tid == 0) {
    unsigned c = sb_below; int bb = 0;
    for (; bb < HBINS - 1; ++bb) { if (c + hist[bb] > (unsigned)RANK_K) break; c += hist[bb]; }
    kth[b] = lo + bb * (w / HBINS);
  }

  // exact global top-20 from the 5120 local candidates
  for (int it = 0; it < TOPF; ++it) {
    float bv = -3.0e38f; int bp = 0;
    for (int j = tid; j < 256 * TOPF; j += 256) if (stv[j] > bv) { bv = stv[j]; bp = j; }
    rv[tid] = bv; rp[tid] = bp; __syncthreads();
    for (int st = 128; st > 0; st >>= 1) {
      if (tid < st && rv[tid + st] > rv[tid]) { rv[tid] = rv[tid + st]; rp[tid] = rp[tid + st]; }
      __syncthreads();
    }
    if (tid == 0) { int p = rp[0]; topv[b * TOPF + it] = rv[0]; topi[b * TOPF + it] = sti[p]; stv[p] = -3.0e38f; }
    __syncthreads();
  }
}

// ---------------- per-sample negative-key threshold tau (2500th largest key) ----------------
__global__ __launch_bounds__(256) void tau_kernel(const float* __restrict__ sim, const float* __restrict__ ru,
                                                  const float* __restrict__ topv, const int* __restrict__ topi,
                                                  const int* __restrict__ bidx, const float* __restrict__ kth,
                                                  float* __restrict__ tau) {
  int b = blockIdx.x, tid = threadIdx.x;
  __shared__ unsigned hist[HBINS];
  __shared__ int excl[TOPF + 1]; __shared__ int nexcl;
  __shared__ int sb_bin; __shared__ unsigned sb_above;
  if (tid == 0) {
    int pos = bidx[b]; int c = 0; excl[c++] = pos;
    for (int j = 0; j < TOPF; ++j) {
      float v = topv[b * TOPF + j]; int i = topi[b * TOPF + j];
      if (v > 0.3f && i != pos) excl[c++] = i;
    }
    nexcl = c;
  }
  for (int i = tid; i < HBINS; i += 256) hist[i] = 0;
  __syncthreads();
  float kthv = kth[b];
  const float* srow = sim + (size_t)b * N_DATA;
  const float* rrow = ru  + (size_t)b * N_DATA;
  int ne = nexcl;

  for (int n = tid; n < N_DATA; n += 256) {
    if (srow[n] >= kthv) {
      bool ex = false;
      for (int j = 0; j < ne; ++j) ex |= (excl[j] == n);
      if (!ex) {
        float u = rrow[n];
        int bin = (int)(u * (float)HBINS);
        bin = bin < 0 ? 0 : (bin > HBINS - 1 ? HBINS - 1 : bin);
        atomicAdd(&hist[bin], 1u);
      }
    }
  }
  __syncthreads();
  if (tid == 0) {
    unsigned c = 0; int bb = HBINS - 1;
    for (; bb > 0; --bb) { if (c + hist[bb] >= KNEG) break; c += hist[bb]; }
    sb_bin = bb; sb_above = c;
  }
  __syncthreads();
  float w  = 1.0f / HBINS;
  float lo = sb_bin * w;
  for (int i = tid; i < HBINS; i += 256) hist[i] = 0;
  __syncthreads();
  for (int n = tid; n < N_DATA; n += 256) {
    if (srow[n] >= kthv) {
      bool ex = false;
      for (int j = 0; j < ne; ++j) ex |= (excl[j] == n);
      if (!ex) {
        float u = rrow[n];
        if (u >= lo && u < lo + w) {
          int sb = (int)((u - lo) * ((float)HBINS / w));
          sb = sb < 0 ? 0 : (sb > HBINS - 1 ? HBINS - 1 : sb);
          atomicAdd(&hist[sb], 1u);
        }
      }
    }
  }
  __syncthreads();
  if (tid == 0) {
    unsigned c = sb_above; int bb = HBINS - 1;
    for (; bb > 0; --bb) { if (c + hist[bb] >= KNEG) break; c += hist[bb]; }
    tau[b] = lo + bb * (w / HBINS);
  }
}

// ---------------- per-sample loss ----------------
__global__ __launch_bounds__(256) void loss_kernel(const float* __restrict__ sim, const float* __restrict__ ru,
                                                   const float* __restrict__ mem, const float* __restrict__ anchor,
                                                   const float* __restrict__ topv, const int* __restrict__ topi,
                                                   const int* __restrict__ bidx, const float* __restrict__ kth,
                                                   const float* __restrict__ tau, float* __restrict__ losses) {
  int b = blockIdx.x, tid = threadIdx.x;
  __shared__ float anch[BIT];
  __shared__ int excl[TOPF + 1]; __shared__ int nexcl;
  __shared__ float red[256];
  __shared__ float term[TOPF + 1];
  if (tid < BIT) anch[tid] = anchor[b * BIT + tid];
  if (tid == 0) {
    int pos = bidx[b]; int c = 0; excl[c++] = pos;
    for (int j = 0; j < TOPF; ++j) {
      float v = topv[b * TOPF + j]; int i = topi[b * TOPF + j];
      if (v > 0.3f && i != pos) excl[c++] = i;
    }
    nexcl = c;
  }
  __syncthreads();
  float kthv = kth[b], tauv = tau[b];
  const float* srow = sim + (size_t)b * N_DATA;
  const float* rrow = ru  + (size_t)b * N_DATA;
  int ne = nexcl;

  float acc = 0.0f;
  for (int n = tid; n < N_DATA; n += 256) {
    if (srow[n] >= kthv && rrow[n] >= tauv) {
      bool ex = false;
      for (int j = 0; j < ne; ++j) ex |= (excl[j] == n);
      if (!ex) {
        const float* mr = mem + (size_t)n * BIT;
        float d = 0.0f;
        #pragma unroll
        for (int k = 0; k < BIT; ++k) d += mr[k] * anch[k];
        acc += expf(d * SC);     // sign(mem)·anchor / T == 8·(mem·anchor)/T
      }
    }
  }
  red[tid] = acc; __syncthreads();
  for (int st = 128; st > 0; st >>= 1) { if (tid < st) red[tid] += red[tid + st]; __syncthreads(); }

  if (tid <= TOPF) {
    int pos = bidx[b];
    int idx = (tid == 0) ? pos : topi[b * TOPF + tid - 1];
    const float* mr = mem + (size_t)idx * BIT;
    float d = 0.0f;
    #pragma unroll
    for (int k = 0; k < BIT; ++k) d += mr[k] * anch[k];
    float s = d * SC;
    if (tid == 0) term[0] = expf(s);
    else {
      float v = topv[b * TOPF + tid - 1];
      bool valid = (v > 0.3f) && (idx != pos);
      term[tid] = valid ? s * expf(s) : 0.0f;
    }
  }
  __syncthreads();
  if (tid == 0) {
    int pos = bidx[b];
    float pos_exp = term[0];
    float numer = pos_exp, nf = 0.0f;
    for (int j = 0; j < TOPF; ++j) {
      float v = topv[b * TOPF + j]; int i = topi[b * TOPF + j];
      if (v > 0.3f && i != pos) { numer += term[j + 1]; nf += 1.0f; }
    }
    float denom = red[0] + pos_exp;
    losses[b] = -logf(numer / denom) / (1.0f + nf);
  }
}

__global__ void mean_kernel(const float* __restrict__ losses, float* __restrict__ out) {
  __shared__ float s[BATCH];
  int t = threadIdx.x;
  s[t] = losses[t]; __syncthreads();
  for (int st = 128; st > 0; st >>= 1) { if (t < st) s[t] += s[t + st]; __syncthreads(); }
  if (t == 0) out[0] = s[0] / (float)BATCH;
}

extern "C" void kernel_launch(void* const* d_in, const int* in_sizes, int n_in,
                              void* d_out, int out_size, void* d_ws, size_t ws_size,
                              hipStream_t stream) {
  const float* iA   = (const float*)d_in[0];
  const float* iB   = (const float*)d_in[1];
  const float* tA   = (const float*)d_in[2];
  const float* tB   = (const float*)d_in[3];
  const float* mem  = (const float*)d_in[4];
  const float* ru   = (const float*)d_in[5];
  const int*   bidx = (const int*)d_in[6];
  float* out = (float*)d_out;

  char* ws = (char*)d_ws;
  size_t off = 0;
  float* sim     = (float*)(ws + off); off += (size_t)BATCH * N_DATA * sizeof(float);
  float* fsum    = (float*)(ws + off); off += BATCH * BIT * sizeof(float);
  float* anchor  = (float*)(ws + off); off += BATCH * BIT * sizeof(float);
  float* newfeat = (float*)(ws + off); off += BATCH * BIT * sizeof(float);
  float* topv    = (float*)(ws + off); off += BATCH * TOPF * sizeof(float);
  int*   topi    = (int*)  (ws + off); off += BATCH * TOPF * sizeof(int);
  float* kth     = (float*)(ws + off); off += 1024;
  float* tau     = (float*)(ws + off); off += 1024;
  float* losses  = (float*)(ws + off); off += 1024;

  // new_memory = copy of memory, then overwrite touched rows
  hipMemcpyAsync(out + 1, mem, (size_t)N_DATA * BIT * sizeof(float),
                 hipMemcpyDeviceToDevice, stream);
  fuse_kernel  <<<BATCH, BIT, 0, stream>>>(iA, iB, tA, tB, fsum, anchor, newfeat);
  update_kernel<<<BATCH, BIT, 0, stream>>>(mem, newfeat, bidx, out + 1);
  gemm_kernel  <<<25000, 256, 0, stream>>>(fsum, mem, sim);
  topk_kth_kernel<<<BATCH, 256, 0, stream>>>(sim, topv, topi, kth);
  tau_kernel   <<<BATCH, 256, 0, stream>>>(sim, ru, topv, topi, bidx, kth, tau);
  loss_kernel  <<<BATCH, 256, 0, stream>>>(sim, ru, mem, anchor, topv, topi, bidx, kth, tau, losses);
  mean_kernel  <<<1, BATCH, 0, stream>>>(losses, out);
}
